// DTNNStep_76063870812667
// MI455X (gfx1250) — matrix operand order repaired and checked by reading, not compiled, not run on hardware
//
#include <hip/hip_runtime.h>
#include <math.h>

// ---- problem sizes (match reference) ----
#define N_ATOMS 100000
#define N_PAIRS 2000000
#define N_EMB   30
#define N_DIST  100
#define N_HID   60

// ---- padded sizes for WMMA tiling ----
#define HP   64    // hidden padded (60 -> 64), K of GEMM2, N of GEMM1
#define KD   128   // distance K padded (100 -> 128)
#define FPAD 32    // output features padded (30 -> 32)

#define TILES (N_PAIRS / 16)   // 125000 edge tiles of 16
#define WPB   8                // waves per block
#define NBLK  2048             // persistent blocks (grid-stride over tiles)

typedef __attribute__((ext_vector_type(16))) __bf16 v16bf;
typedef __attribute__((ext_vector_type(8)))  float  v8f;
typedef __attribute__((ext_vector_type(2)))  __bf16 bf16x2;

union FragBF { unsigned u[8]; v16bf v; };

// native bf16 converts (fptrunc -> v_cvt_(pk_)bf16_f32, RNE)
__device__ __forceinline__ unsigned pack2bf(float a, float b) {
    bf16x2 p; p.x = (__bf16)a; p.y = (__bf16)b;
    return __builtin_bit_cast(unsigned, p);
}
__device__ __forceinline__ unsigned short f2bf(float f) {
    __bf16 h = (__bf16)f;
    return __builtin_bit_cast(unsigned short, h);
}

// fast tanh: copysign(1 - 2/(e^(2|x|)+1), x); exp/rcp ride the TRANS pipe
__device__ __forceinline__ float fast_tanh(float x) {
    float e = __expf(2.0f * fabsf(x));
    float r = 1.0f - __fdividef(2.0f, e + 1.0f);
    return copysignf(r, x);
}

// A-matrix (16x32 bf16) per-VGPR K offset within a 32-wide K block (contiguous: k = 16g + 2v)
__device__ __forceinline__ int a_koff(int v, int g) {
    return 2 * v + g * 16;
}
// float4 start covering VGPR pair (2p, 2p+1) within a 32-wide K block (contiguous: k = 16g + 4p)
__device__ __forceinline__ int a_koff4(int p, int g) {
    return 4 * p + g * 16;
}

// ---------------------------------------------------------------------------
// Prep 1: atom_hidden = atom_features @ W_cf + b_cf, padded to [N_ATOMS][64]
// ---------------------------------------------------------------------------
__global__ void prep_atom_hidden(const float* __restrict__ af,
                                 const float* __restrict__ Wcf,
                                 const float* __restrict__ bcf,
                                 float* __restrict__ ah) {
    int idx = blockIdx.x * blockDim.x + threadIdx.x;
    if (idx >= N_ATOMS * HP) return;
    int n = idx >> 6, h = idx & 63;
    float s = 0.f;
    if (h < N_HID) {
        s = bcf[h];
        const float* a = af + (long)n * N_EMB;
        #pragma unroll
        for (int k = 0; k < N_EMB; ++k) s += a[k] * Wcf[k * N_HID + h];
    }
    ah[idx] = s;
}

// ---------------------------------------------------------------------------
// Prep 2: transposed, zero-padded bf16 weights:
//   wdf_t: [HP][KD] from W_df[100][60];  wfc_t: [FPAD][HP] from W_fc[60][30]
// ---------------------------------------------------------------------------
__global__ void prep_weights(const float* __restrict__ Wdf,
                             const float* __restrict__ Wfc,
                             unsigned short* __restrict__ wdf_t,
                             unsigned short* __restrict__ wfc_t) {
    int i = blockIdx.x * blockDim.x + threadIdx.x;
    if (i < HP * KD) {
        int n = i >> 7, k = i & 127;
        float v = (n < N_HID && k < N_DIST) ? Wdf[k * N_HID + n] : 0.f;
        wdf_t[i] = f2bf(v);
    } else if (i < HP * KD + FPAD * HP) {
        int j = i - HP * KD;
        int n = j >> 6, k = j & 63;
        float v = (n < N_EMB && k < N_HID) ? Wfc[k * N_EMB + n] : 0.f;
        wfc_t[j] = f2bf(v);
    }
}

// ---------------------------------------------------------------------------
// Prep 3: d_out = atom_features - tanh((b_df * atom_hidden) @ W_fc)
// ---------------------------------------------------------------------------
__global__ void prep_out_init(const float* __restrict__ af,
                              const float* __restrict__ ah,
                              const float* __restrict__ Wfc,
                              const float* __restrict__ bdf,
                              float* __restrict__ out) {
    int idx = blockIdx.x * blockDim.x + threadIdx.x;
    if (idx >= N_ATOMS * 32) return;
    int n = idx >> 5, f = idx & 31;
    if (f >= N_EMB) return;
    const float* a = ah + (long)n * HP;
    float s = 0.f;
    #pragma unroll 4
    for (int h = 0; h < N_HID; ++h) s += (bdf[h] * a[h]) * Wfc[h * N_EMB + f];
    out[(long)n * N_EMB + f] = af[(long)n * N_EMB + f] - fast_tanh(s);
}

// ---------------------------------------------------------------------------
// Main fused edge kernel: persistent waves, one 16-edge tile per iteration.
//   GEMM1 (bf16 WMMA): dist[16x128] @ wdf_t -> dh[16x64] (f32 acc)
//   epilogue: (dh + b_df) * atom_hidden[j]  -> LDS [16x64] bf16
//   GEMM2 (bf16 WMMA): [16x64] @ wfc_t -> [16x32], tanh, atomic scatter
// ---------------------------------------------------------------------------
__global__ void __launch_bounds__(256)
dtnn_edge_kernel(const float* __restrict__ dist,
                 const int*   __restrict__ mem_i,
                 const int*   __restrict__ mem_j,
                 const float* __restrict__ ah,
                 const float* __restrict__ bdf,
                 const unsigned* __restrict__ wdf_t,  // [HP][KD/2] packed bf16x2
                 const unsigned* __restrict__ wfc_t,  // [FPAD][HP/2] packed bf16x2
                 float* __restrict__ out) {
    __shared__ unsigned lds[WPB][16 * (HP / 2)];  // per-wave 16x64 bf16 tile

    const int lane = threadIdx.x & 31;
    const int wib  = threadIdx.x >> 5;
    const int m16  = lane & 15;
    const int g    = lane >> 4;
    const int nWaves = gridDim.x * WPB;

    // hoist GEMM2 weight fragments out of the tile loop (32 VGPRs, reused)
    FragBF bfc[2][2];
    #pragma unroll
    for (int n2 = 0; n2 < 2; ++n2)
        #pragma unroll
        for (int kb = 0; kb < 2; ++kb)
            #pragma unroll
            for (int v = 0; v < 8; ++v)
                bfc[n2][kb].u[v] = wfc_t[(n2 * 16 + m16) * (HP / 2) + kb * 16 + g * 8 + v];

    unsigned short* myrow = (unsigned short*)&lds[wib][0];   // [16][HP] bf16

    for (int tile = blockIdx.x * WPB + wib; tile < TILES; tile += nWaves) {
        const long eBase = (long)tile * 16;

        // membership for this tile's rows (row m16 per lane)
        const int im = mem_i[eBase + m16];
        const int jm = mem_j[eBase + m16];

        // prefetch this wave's next tile's distance rows (global_prefetch_b8)
        if (tile + nWaves < TILES)
            __builtin_prefetch(dist + ((long)(tile + nWaves) * 16 + m16) * N_DIST, 0, 1);

        // ---- A fragments: distance tile fp32 -> bf16, float4 loads ----
        FragBF afr[4];
        #pragma unroll
        for (int kb = 0; kb < 4; ++kb) {
            #pragma unroll
            for (int p = 0; p < 4; ++p) {          // covers VGPRs 2p, 2p+1
                int k = kb * 32 + a_koff4(p, g);   // multiple of 4
                unsigned lo = 0u, hi = 0u;
                if (k < N_DIST) {                  // k<=96 -> k..k+3 in bounds
                    float4 d = *(const float4*)(dist + (eBase + m16) * N_DIST + k);
                    lo = pack2bf(d.x, d.y);
                    hi = pack2bf(d.z, d.w);
                }
                afr[kb].u[2 * p]     = lo;
                afr[kb].u[2 * p + 1] = hi;
            }
        }

        // ---- GEMM1 + fused epilogue per 16-wide hidden column tile ----
        #pragma unroll
        for (int nt = 0; nt < 4; ++nt) {
            v8f c = {0.f, 0.f, 0.f, 0.f, 0.f, 0.f, 0.f, 0.f};
            #pragma unroll
            for (int kb = 0; kb < 4; ++kb) {
                FragBF bf;
                #pragma unroll
                for (int v = 0; v < 8; ++v)        // B: n=m16, K contiguous per half-wave
                    bf.u[v] = wdf_t[(nt * 16 + m16) * (KD / 2) + kb * 16 + g * 8 + v];
                c = __builtin_amdgcn_wmma_f32_16x16x32_bf16(
                        false, afr[kb].v, false, bf.v, (short)0, c, false, false);
            }
            const int h = nt * 16 + m16;
            const float bias = (h < N_HID) ? bdf[h] : 0.f;   // pad cols stay 0
            #pragma unroll
            for (int r = 0; r < 8; ++r) {          // C: element (m = r+8g, n = m16)
                int m = r + 8 * g;
                int j = __shfl(jm, m, 32);         // full EXEC here
                float prod = (c[r] + bias) * ah[(long)j * HP + h];  // L2 gather
                myrow[m * HP + h] = f2bf(prod);
            }
        }

        // ---- A2 fragments from LDS (re-layout C -> A across lanes) ----
        FragBF a2[2];
        #pragma unroll
        for (int kb = 0; kb < 2; ++kb)
            #pragma unroll
            for (int v = 0; v < 8; ++v) {
                int k = kb * 32 + a_koff(v, g);
                a2[kb].u[v] = lds[wib][m16 * (HP / 2) + (k >> 1)];
            }

        // ---- GEMM2 + tanh + atomic scatter ----
        #pragma unroll
        for (int n2 = 0; n2 < 2; ++n2) {
            v8f c2 = {0.f, 0.f, 0.f, 0.f, 0.f, 0.f, 0.f, 0.f};
            #pragma unroll
            for (int kb = 0; kb < 2; ++kb)
                c2 = __builtin_amdgcn_wmma_f32_16x16x32_bf16(
                        false, a2[kb].v, false, bfc[n2][kb].v, (short)0, c2, false, false);
            const int f = n2 * 16 + m16;
            #pragma unroll
            for (int r = 0; r < 8; ++r) {
                int m = r + 8 * g;
                int i = __shfl(im, m, 32);         // shfl before divergence
                if (f < N_EMB)
                    atomicAdd(out + (long)i * N_EMB + f, fast_tanh(c2[r]));
            }
        }
    }
}

// ---------------------------------------------------------------------------
extern "C" void kernel_launch(void* const* d_in, const int* in_sizes, int n_in,
                              void* d_out, int out_size, void* d_ws, size_t ws_size,
                              hipStream_t stream) {
    const float* af   = (const float*)d_in[0];
    const float* dist = (const float*)d_in[1];
    const int*   mi   = (const int*)d_in[2];
    const int*   mj   = (const int*)d_in[3];
    const float* Wcf  = (const float*)d_in[4];
    const float* Wdf  = (const float*)d_in[5];
    const float* Wfc  = (const float*)d_in[6];
    const float* bcf  = (const float*)d_in[7];
    const float* bdf  = (const float*)d_in[8];
    float* out = (float*)d_out;

    // workspace: padded atom_hidden (25.6 MB) + bf16 weight tiles (~20.5 KB)
    char* ws = (char*)d_ws;
    float* ah = (float*)ws;
    size_t off = (size_t)N_ATOMS * HP * sizeof(float);
    unsigned short* wdf_t = (unsigned short*)(ws + off); off += (size_t)HP * KD * 2;
    unsigned short* wfc_t = (unsigned short*)(ws + off);

    prep_atom_hidden<<<(N_ATOMS * HP + 255) / 256, 256, 0, stream>>>(af, Wcf, bcf, ah);
    prep_weights<<<(HP * KD + FPAD * HP + 255) / 256, 256, 0, stream>>>(Wdf, Wfc, wdf_t, wfc_t);
    prep_out_init<<<(N_ATOMS * 32 + 255) / 256, 256, 0, stream>>>(af, ah, Wfc, bdf, out);

    dtnn_edge_kernel<<<NBLK, 256, 0, stream>>>(
        dist, mi, mj, ah, bdf,
        (const unsigned*)wdf_t, (const unsigned*)wfc_t, out);
}
